// GraphAttentionNetwork_12601434047043
// MI455X (gfx1250) — matrix operand order repaired and checked by reading, not compiled, not run on hardware
//
#include <hip/hip_runtime.h>
#include <hip/hip_bf16.h>
#include <math.h>

// ---------------------------------------------------------------------------
// GAT forward for MI455X (gfx1250, wave32, WMMA).
//
// Sizes: N=4096 nodes, F_IN=256, F_HID=128, H=8 heads, F_OUT=256.
//
// Roofline: att@h aggregations = 43 GFLOP, feature GEMMs = 4.6 GFLOP ->
// v_wmma_f32_16x16x32_bf16. Materializing the 9 NxN attention maps = 576 MB
// (> 192 MB L2) of HBM traffic, so the softmax is fused: leaky_relu is
// monotonic => masked row max = lrelu(s1_i + masked-max(s2)); denominator via
// one exp sweep over adj (adj = 64 MB, L2-resident). The aggregation kernel
// builds P tiles in registers (fast __expf), packs them as WMMA A-fragments,
// and multiplies against prepacked h B-fragments that are double-buffered
// through LDS with GLOBAL_LOAD_ASYNC_TO_LDS_B128 (s_wait_asynccnt).
// ---------------------------------------------------------------------------

#define NNODE 4096
#define FIN   256
#define FHID  128
#define NHEAD 8
#define FOUT  256

typedef __attribute__((ext_vector_type(16))) __bf16   v16bf;
typedef __attribute__((ext_vector_type(2)))  __bf16   v2bf;
typedef __attribute__((ext_vector_type(2)))  float    v2f;
typedef __attribute__((ext_vector_type(8)))  float    v8f;
typedef __attribute__((ext_vector_type(8)))  unsigned v8u;
typedef __attribute__((ext_vector_type(4)))  int      av4i;   // builtin's arg type

#if defined(__has_builtin)
#  if __has_builtin(__builtin_amdgcn_cvt_pk_bf16_f32)
#    define HAVE_CVT_PK_BF16 1
#  endif
#  if __has_builtin(__builtin_amdgcn_global_load_async_to_lds_b128)
#    define HAVE_ASYNC_LDS 1
#  endif
#  if __has_builtin(__builtin_amdgcn_s_wait_asynccnt)
#    define HAVE_WAIT_ASYNC 1
#  endif
#endif
#ifndef HAVE_CVT_PK_BF16
#  define HAVE_CVT_PK_BF16 0
#endif
#ifndef HAVE_ASYNC_LDS
#  define HAVE_ASYNC_LDS 0
#endif
#ifndef HAVE_WAIT_ASYNC
#  define HAVE_WAIT_ASYNC 0
#endif

// pack two floats into one dword of bf16x2 (low = a, high = b)
static __device__ __forceinline__ unsigned pkbf(float a, float b) {
#if HAVE_CVT_PK_BF16
  return __builtin_bit_cast(unsigned, __builtin_amdgcn_cvt_pk_bf16_f32(a, b));
#else
  // fptrunc f32->bf16 (RNE); lowers to the packed hw cvt when available
  v2f t; t.x = a; t.y = b;
  v2bf r = __builtin_convertvector(t, v2bf);
  return __builtin_bit_cast(unsigned, r);
#endif
}

static __device__ __forceinline__ v8f wmma_bf16(v8u a, v8u b, v8f c) {
  // D = A(16x32 bf16) * B(32x16 bf16) + C(16x16 f32)
  return __builtin_amdgcn_wmma_f32_16x16x32_bf16(
      false, __builtin_bit_cast(v16bf, a), false, __builtin_bit_cast(v16bf, b),
      (short)0, c, false, false);
}

static __device__ __forceinline__ void wait_asynccnt0() {
#if HAVE_WAIT_ASYNC
  __builtin_amdgcn_s_wait_asynccnt(0);
#elif HAVE_ASYNC_LDS
  asm volatile("s_wait_asynccnt 0x0" ::: "memory");
#endif
}

#if HAVE_ASYNC_LDS
static __device__ __forceinline__ void async_b128(const unsigned* g, unsigned* l) {
  __builtin_amdgcn_global_load_async_to_lds_b128((av4i*)g, (av4i*)l, 0, 0);
}
#endif

// ---------------------------------------------------------------------------
// Generic bf16 WMMA GEMM: C[M,Nn] = A[M,K] * B[K,Nn], all f32 row-major.
// One wave -> one 16x16 C tile; 4 waves/block cover 64 output columns.
// blockIdx.z selects a slice via the *StrideZ args (used for the 8 heads).
// epilogue: 0 = plain store, 1 = +bias then exact GELU.
//
// Fragment layouts per CDNA5 ISA 7.12.2 (wave32):
//   A 16x32 bf16 : lane l holds row l&15; dword v<4 -> K=(l>>4)*8+2v(+1),
//                  v>=4 -> K=16+(l>>4)*8+2(v-4)(+1)
//   B 32x16 bf16 : lane l holds col l&15; same K mapping as A
//   C/D 16x16 f32: VGPR r -> row r+8*(l>>4), col l&15
// ---------------------------------------------------------------------------
__global__ void gemm_bf16_kernel(const float* __restrict__ A,
                                 const float* __restrict__ B,
                                 float* __restrict__ C,
                                 const float* __restrict__ bias,
                                 int M, int Nn, int K,
                                 long aStrideZ, long bStrideZ, long cStrideZ,
                                 int epilogue) {
  const int lane = threadIdx.x & 31;
  const int wave = threadIdx.x >> 5;
  const int rowBase = blockIdx.x * 16;
  const int colBase = (blockIdx.y * 4 + wave) * 16;
  if (colBase >= Nn) return;                     // whole wave exits together
  const int z = blockIdx.z;
  A += (long)z * aStrideZ;
  B += (long)z * bStrideZ;
  C += (long)z * cStrideZ;

  const int g  = lane >> 4;
  const int rl = lane & 15;
  const float* arow = A + (long)(rowBase + rl) * K;
  const float* bcol = B + colBase + rl;

  v8f acc = {};
  for (int kc = 0; kc < K; kc += 32) {
    const float* p1 = arow + kc + g * 8;         // K run g*8 .. g*8+7
    const float* p2 = p1 + 16;                   // K run 16+g*8 .. +7
    v8u aw, bw;
#pragma unroll
    for (int v = 0; v < 4; ++v) {
      aw[v]     = pkbf(p1[2 * v], p1[2 * v + 1]);
      aw[4 + v] = pkbf(p2[2 * v], p2[2 * v + 1]);
    }
    const float* pb = bcol + (long)(kc + g * 8) * Nn;
#pragma unroll
    for (int v = 0; v < 8; ++v) {
      const int kk = (v < 4) ? 2 * v : 16 + 2 * (v - 4);
      bw[v] = pkbf(pb[(long)kk * Nn], pb[(long)(kk + 1) * Nn]);
    }
    acc = wmma_bf16(aw, bw, acc);
  }

#pragma unroll
  for (int rr = 0; rr < 8; ++rr) {
    const int row = rowBase + rr + 8 * g;
    const int col = colBase + rl;
    float v = acc[rr];
    if (epilogue == 1) {
      v += bias[col];
      v = 0.5f * v * (1.0f + erff(v * 0.70710678118654752f));  // exact GELU
    }
    C[(long)row * Nn + col] = v;
  }
}

// ---------------------------------------------------------------------------
// s1[i] = h[i,:].a[:F], s2[i] = h[i,:].a[F:2F].  One 128-thread block per row.
// ---------------------------------------------------------------------------
__global__ void rowdots_kernel(const float* __restrict__ h,
                               const float* __restrict__ a,
                               float* __restrict__ s1, float* __restrict__ s2,
                               int F, long hStrideZ, long aStrideZ) {
  const int row = blockIdx.x, z = blockIdx.y;
  h += (long)z * hStrideZ + (long)row * F;
  a += (long)z * aStrideZ;
  __shared__ float r1[128], r2[128];
  float p1 = 0.f, p2 = 0.f;
  for (int f = threadIdx.x; f < F; f += 128) {
    const float v = h[f];
    p1 += v * a[f];
    p2 += v * a[F + f];
  }
  r1[threadIdx.x] = p1; r2[threadIdx.x] = p2;
  __syncthreads();
  for (int s = 64; s > 0; s >>= 1) {
    if ((int)threadIdx.x < s) {
      r1[threadIdx.x] += r1[threadIdx.x + s];
      r2[threadIdx.x] += r2[threadIdx.x + s];
    }
    __syncthreads();
  }
  if (threadIdx.x == 0) {
    s1[(long)z * NNODE + row] = r1[0];
    s2[(long)z * NNODE + row] = r2[0];
  }
}

// ---------------------------------------------------------------------------
// Masked softmax stats per (row, head):
//   m_i  = lrelu(s1_i + max_{adj>0} s2_j)      (lrelu monotonic)
//   di_i = 1 / sum_{adj>0} exp(lrelu(s1_i+s2_j) - m_i)
// ---------------------------------------------------------------------------
__global__ void attn_stats_kernel(const int* __restrict__ adj,
                                  const float* __restrict__ s1,
                                  const float* __restrict__ s2,
                                  float* __restrict__ m,
                                  float* __restrict__ di) {
  const int row = blockIdx.x, z = blockIdx.y;
  const int* ar = adj + (long)row * NNODE;
  s1 += (long)z * NNODE; s2 += (long)z * NNODE;
  __shared__ float red[256];

  float lm = -3.0e38f;
  for (int j = threadIdx.x; j < NNODE; j += 256)
    if (ar[j] > 0) lm = fmaxf(lm, s2[j]);
  red[threadIdx.x] = lm;
  __syncthreads();
  for (int s = 128; s > 0; s >>= 1) {
    if ((int)threadIdx.x < s)
      red[threadIdx.x] = fmaxf(red[threadIdx.x], red[threadIdx.x + s]);
    __syncthreads();
  }
  const float s1i = s1[row];
  float mm = s1i + red[0];
  mm = mm > 0.f ? mm : 0.01f * mm;
  __syncthreads();

  float ls = 0.f;
  for (int j = threadIdx.x; j < NNODE; j += 256)
    if (ar[j] > 0) {
      float e = s1i + s2[j];
      e = e > 0.f ? e : 0.01f * e;
      ls += __expf(e - mm);
    }
  red[threadIdx.x] = ls;
  __syncthreads();
  for (int s = 128; s > 0; s >>= 1) {
    if ((int)threadIdx.x < s) red[threadIdx.x] += red[threadIdx.x + s];
    __syncthreads();
  }
  if (threadIdx.x == 0) {
    m [(long)z * NNODE + row] = mm;
    di[(long)z * NNODE + row] = red[0] > 0.f ? 1.0f / red[0] : 0.0f;
  }
}

// ---------------------------------------------------------------------------
// Pack h (f32 [rows,F]) into WMMA B-fragment order so the aggregation kernel
// loads each fragment as 8 contiguous dwords per lane (2x b128).
// dword index = ((c*nT + t)*32 + lane)*8 + v,  value = bf16x2 of
//   h[c*32 + (v<4 ? (lane>>4)*8 + 2v : 16 + (lane>>4)*8 + 2(v-4)) (+1)][t*16 + (lane&15)]
// ---------------------------------------------------------------------------
__global__ void pack_b_kernel(const float* __restrict__ src,
                              unsigned* __restrict__ dst,
                              int F, long srcStrideZ, long dstStrideZ) {
  long flat = (long)blockIdx.x * blockDim.x + threadIdx.x;
  const int z = blockIdx.y;
  src += (long)z * srcStrideZ;
  dst += (long)z * dstStrideZ;
  const int v    = (int)(flat & 7);
  const int lane = (int)((flat >> 3) & 31);
  const long ct  = flat >> 8;          // c*nT + t
  const int nT   = F >> 4;
  const long c   = ct / nT;
  const int  t   = (int)(ct % nT);
  const int  g   = lane >> 4;
  const int  col = t * 16 + (lane & 15);
  const long j0  = c * 32 + (v < 4 ? g * 8 + 2 * v : 16 + g * 8 + 2 * (v - 4));
  dst[flat] = pkbf(src[j0 * F + col], src[(j0 + 1) * F + col]);
}

// ---------------------------------------------------------------------------
// Fused masked-softmax aggregation + ELU:  xout tile = elu(P @ h).
// One wave per (16-row tile, blockIdx.y). Per 32-column adj chunk: build P
// as a 16x32 bf16 A-fragment in registers (__expf on VALU), 8 WMMAs against
// prepacked B-fragments double-buffered through LDS with async loads.
// Layer 1: blockIdx.y = head (concat into xout columns h*128+f).
// Layer 2: blockIdx.y = f-block (tBase = 8*y), single head.
// ---------------------------------------------------------------------------
__global__ void attn_agg_kernel(const int* __restrict__ adj,
                                const float* __restrict__ s1b,
                                const float* __restrict__ s2b,
                                const float* __restrict__ mb,
                                const float* __restrict__ dib,
                                const unsigned* __restrict__ pBb,
                                float* __restrict__ xout,
                                int sStrideY, long pbStrideY,
                                int nTtot, int tBaseMulY, int colOffMulY,
                                int outStride) {
  const int lane = threadIdx.x;
  const int y = blockIdx.y;
  const float* s1 = s1b + (long)sStrideY * y;
  const float* s2 = s2b + (long)sStrideY * y;
  const float* mB = mb  + (long)sStrideY * y;
  const float* dB = dib + (long)sStrideY * y;
  const unsigned* pB = pBb + pbStrideY * y;
  const int tBase  = tBaseMulY * y;
  const int colOff = colOffMulY * y;

  const int rowBase = blockIdx.x * 16;
  const int g  = lane >> 4;
  const int rl = lane & 15;
  const int i  = rowBase + rl;
  const int* adjRow = adj + (long)i * NNODE;
  const float s1i = s1[i];
  const float mi  = mB[i];
  const float dii = dB[i];

#if HAVE_ASYNC_LDS
  __shared__ unsigned ldsB[2][8 * 32 * 8];     // 2 x 8 KB double buffer
  // prologue: async-load chunk 0 into buffer 0
  {
    const unsigned* gsrc = pB + lane * 8;
#pragma unroll
    for (int t = 0; t < 8; ++t) {
      const unsigned* gp = gsrc + (long)(tBase + t) * 256;
      unsigned* lp = &ldsB[0][(t * 32 + lane) * 8];
      async_b128(gp, lp);
      async_b128(gp + 4, lp + 4);
    }
  }
  wait_asynccnt0();
#endif

  v8f acc[8] = {};
  for (int c = 0; c < NNODE / 32; ++c) {
#if HAVE_ASYNC_LDS
    if (c + 1 < NNODE / 32) {   // prefetch next chunk into the other buffer
      const unsigned* gsrc = pB + (long)(c + 1) * nTtot * 256 + lane * 8;
#pragma unroll
      for (int t = 0; t < 8; ++t) {
        const unsigned* gp = gsrc + (long)(tBase + t) * 256;
        unsigned* lp = &ldsB[(c + 1) & 1][(t * 32 + lane) * 8];
        async_b128(gp, lp);
        async_b128(gp + 4, lp + 4);
      }
    }
#endif
    const int j1 = c * 32 + g * 8;          // K run 1
    const int j2 = j1 + 16;                 // K run 2
    __builtin_prefetch(adjRow + (c + 1) * 32, 0, 1);   // global_prefetch_b8

    const int4   a0 = *(const int4*)  (adjRow + j1);
    const int4   a1 = *(const int4*)  (adjRow + j1 + 4);
    const int4   a2 = *(const int4*)  (adjRow + j2);
    const int4   a3 = *(const int4*)  (adjRow + j2 + 4);
    const float4 t0 = *(const float4*)(s2 + j1);
    const float4 t1 = *(const float4*)(s2 + j1 + 4);
    const float4 t2 = *(const float4*)(s2 + j2);
    const float4 t3 = *(const float4*)(s2 + j2 + 4);
    const int   aa[16] = {a0.x,a0.y,a0.z,a0.w, a1.x,a1.y,a1.z,a1.w,
                          a2.x,a2.y,a2.z,a2.w, a3.x,a3.y,a3.z,a3.w};
    const float ss[16] = {t0.x,t0.y,t0.z,t0.w, t1.x,t1.y,t1.z,t1.w,
                          t2.x,t2.y,t2.z,t2.w, t3.x,t3.y,t3.z,t3.w};

    float p[16];
#pragma unroll
    for (int u = 0; u < 16; ++u) {
      float e = s1i + ss[u];
      e = e > 0.f ? e : 0.01f * e;
      p[u] = (aa[u] > 0) ? __expf(e - mi) * dii : 0.0f;
    }
    v8u aw;
#pragma unroll
    for (int v = 0; v < 8; ++v) aw[v] = pkbf(p[2 * v], p[2 * v + 1]);

#if HAVE_ASYNC_LDS
#pragma unroll
    for (int t = 0; t < 8; ++t) {
      const v8u bu = *(const v8u*)&ldsB[c & 1][(t * 32 + lane) * 8];
      acc[t] = wmma_bf16(aw, bu, acc[t]);
    }
    wait_asynccnt0();   // chunk c+1 landed before next iteration reads it
#else
    const unsigned* pbc = pB + (long)c * nTtot * 256 + lane * 8;
#pragma unroll
    for (int t = 0; t < 8; ++t) {
      const v8u bu = *(const v8u*)(pbc + (long)(tBase + t) * 256);
      acc[t] = wmma_bf16(aw, bu, acc[t]);
    }
#endif
  }

#pragma unroll
  for (int t = 0; t < 8; ++t) {
#pragma unroll
    for (int rr = 0; rr < 8; ++rr) {
      float v = acc[t][rr];
      v = v > 0.f ? v : __expf(v) - 1.0f;              // ELU (alpha=1)
      const int orow = rowBase + rr + 8 * g;
      const int ocol = colOff + (tBase + t) * 16 + rl;
      xout[(long)orow * outStride + ocol] = v;
    }
  }
}

// ---------------------------------------------------------------------------
// Final fc2: y[i] = y1[i,:].w + b  (128-length dot per row)
// ---------------------------------------------------------------------------
__global__ void fc2_kernel(const float* __restrict__ y1,
                           const float* __restrict__ w,
                           const float* __restrict__ b,
                           float* __restrict__ out) {
  const int row = blockIdx.x;
  __shared__ float red[128];
  red[threadIdx.x] = y1[(long)row * FHID + threadIdx.x] * w[threadIdx.x];
  __syncthreads();
  for (int s = 64; s > 0; s >>= 1) {
    if ((int)threadIdx.x < s) red[threadIdx.x] += red[threadIdx.x + s];
    __syncthreads();
  }
  if (threadIdx.x == 0) out[row] = red[0] + b[0];
}

// ---------------------------------------------------------------------------
extern "C" void kernel_launch(void* const* d_in, const int* in_sizes, int n_in,
                              void* d_out, int out_size, void* d_ws, size_t ws_size,
                              hipStream_t stream) {
  const float* x    = (const float*)d_in[0];
  const int*   adj  = (const int*)  d_in[1];
  const float* Wh   = (const float*)d_in[2];   // [H,256,128]
  const float* ah   = (const float*)d_in[3];   // [H,256]
  const float* Wo   = (const float*)d_in[4];   // [1024,256]
  const float* ao   = (const float*)d_in[5];   // [512]
  const float* fc1w = (const float*)d_in[6];   // [256,128]
  const float* fc1b = (const float*)d_in[7];   // [128]
  const float* fc2w = (const float*)d_in[8];   // [128,1]
  const float* fc2b = (const float*)d_in[9];   // [1]
  float* out = (float*)d_out;

  char* wsp = (char*)d_ws;
  auto alloc = [&](size_t bytes) -> char* {
    char* p = wsp;
    wsp += (bytes + 255) & ~(size_t)255;
    return p;
  };
  float*    h1   = (float*)   alloc((size_t)NHEAD * NNODE * FHID * 4);   // 16 MB
  unsigned* pB1  = (unsigned*)alloc((size_t)NHEAD * NNODE * FHID * 2);   //  8 MB
  float*    s1h  = (float*)   alloc((size_t)NHEAD * NNODE * 4);
  float*    s2h  = (float*)   alloc((size_t)NHEAD * NNODE * 4);
  float*    m1   = (float*)   alloc((size_t)NHEAD * NNODE * 4);
  float*    di1  = (float*)   alloc((size_t)NHEAD * NNODE * 4);
  float*    xcat = (float*)   alloc((size_t)NNODE * (NHEAD * FHID) * 4); // 16 MB
  float*    h2   = (float*)   alloc((size_t)NNODE * FOUT * 4);
  unsigned* pB2  = (unsigned*)alloc((size_t)NNODE * FOUT * 2);
  float*    s1o  = (float*)   alloc((size_t)NNODE * 4);
  float*    s2o  = (float*)   alloc((size_t)NNODE * 4);
  float*    m2   = (float*)   alloc((size_t)NNODE * 4);
  float*    di2  = (float*)   alloc((size_t)NNODE * 4);
  float*    x2   = (float*)   alloc((size_t)NNODE * FOUT * 4);
  float*    y1   = (float*)   alloc((size_t)NNODE * FHID * 4);

  // ---- Layer 1: h1[h] = x @ W_heads[h]  (8 heads via grid.z) ----
  gemm_bf16_kernel<<<dim3(NNODE / 16, FHID / 64, NHEAD), 128, 0, stream>>>(
      x, Wh, h1, nullptr, NNODE, FHID, FIN,
      0, (long)FIN * FHID, (long)NNODE * FHID, 0);

  rowdots_kernel<<<dim3(NNODE, NHEAD), 128, 0, stream>>>(
      h1, ah, s1h, s2h, FHID, (long)NNODE * FHID, (long)2 * FHID);

  attn_stats_kernel<<<dim3(NNODE, NHEAD), 256, 0, stream>>>(
      adj, s1h, s2h, m1, di1);

  pack_b_kernel<<<dim3((NNODE * FHID / 2) / 256, NHEAD), 256, 0, stream>>>(
      h1, pB1, FHID, (long)NNODE * FHID, (long)NNODE * FHID / 2);

  attn_agg_kernel<<<dim3(NNODE / 16, NHEAD), 32, 0, stream>>>(
      adj, s1h, s2h, m1, di1, pB1, xcat,
      NNODE, (long)NNODE * FHID / 2, FHID / 16, 0, FHID, NHEAD * FHID);

  // ---- Layer 2: h2 = xcat @ W_out ----
  gemm_bf16_kernel<<<dim3(NNODE / 16, FOUT / 64, 1), 128, 0, stream>>>(
      xcat, Wo, h2, nullptr, NNODE, FOUT, NHEAD * FHID, 0, 0, 0, 0);

  rowdots_kernel<<<dim3(NNODE, 1), 128, 0, stream>>>(
      h2, ao, s1o, s2o, FOUT, 0, 0);

  attn_stats_kernel<<<dim3(NNODE, 1), 256, 0, stream>>>(
      adj, s1o, s2o, m2, di2);

  pack_b_kernel<<<dim3((NNODE * FOUT / 2) / 256, 1), 256, 0, stream>>>(
      h2, pB2, FOUT, 0, 0);

  attn_agg_kernel<<<dim3(NNODE / 16, 2), 32, 0, stream>>>(
      adj, s1o, s2o, m2, di2, pB2, x2,
      0, 0, FOUT / 16, 8, 0, FOUT);

  // ---- MLP head: y = (gelu(x2 @ fc1w + b1)) @ fc2w + b2 ----
  gemm_bf16_kernel<<<dim3(NNODE / 16, FHID / 64, 1), 128, 0, stream>>>(
      x2, fc1w, y1, fc1b, NNODE, FHID, FIN, 0, 0, 0, 1);

  fc2_kernel<<<dim3(NNODE), 128, 0, stream>>>(y1, fc2w, fc2b, out);
}
